// Tacotron2_44513041055825
// MI455X (gfx1250) — matrix operand order repaired and checked by reading, not compiled, not run on hardware
//
#include <hip/hip_runtime.h>

typedef unsigned short u16;
typedef __attribute__((ext_vector_type(16))) __bf16 v16bf;
typedef __attribute__((ext_vector_type(8)))  float  v8f;

#define BB 32
#define TE 200
#define TD 800
#define NMEL 80
#define NSTEP 400
#define NB 16
#define TPB 256
#define HAL 64
#define THE (TE + 2 * HAL)   // encoder conv time stride (328)
#define THP (TD + 2 * HAL)   // postnet conv time stride (928)

union BFV { v16bf v; u16 u[16]; };

__device__ __forceinline__ u16 f2bf(float f) {
    unsigned int u = __float_as_uint(f);
    u += 0x7fffu + ((u >> 16) & 1u);
    return (u16)(u >> 16);
}
__device__ __forceinline__ float sigf(float x) { return 1.f / (1.f + __expf(-x)); }
__device__ __forceinline__ float actf(float x, int a) {
    if (a == 1) return fmaxf(x, 0.f);
    if (a == 2) return tanhf(x);
    if (a == 3) return sigf(x);
    return x;
}

// ---------- WMMA 16x16x32 bf16 fragment helpers (ISA 7.12.2 layouts) ----------
// A: 16(m) x 32(k), element (m,k) = A[m*ars + k]
// B: 16(n) x 32(k), element (n,k) = B[n*brs + k]   (weight layout = math B^T)
__device__ __forceinline__ v16bf ldfa(const u16* A, long ars)
{
    int lane = threadIdx.x & 31;
    int grp = lane >> 4, mr = lane & 15;
    BFV a;
#pragma unroll
    for (int v = 0; v < 8; ++v) {
        int kb = v * 2 + (v >= 4 ? 8 : 0) + grp * 8;
        a.u[2 * v]     = A[(long)mr * ars + kb];
        a.u[2 * v + 1] = A[(long)mr * ars + kb + 1];
    }
    return a.v;
}
__device__ __forceinline__ v16bf ldfb(const u16* B, long brs)
{
    int lane = threadIdx.x & 31;
    int grp = lane >> 4, mr = lane & 15;
    BFV b;
#pragma unroll
    for (int e = 0; e < 16; ++e)
        b.u[e] = B[(long)mr * brs + grp * 8 + e + (e >= 8 ? 8 : 0)];
    return b.v;
}
__device__ __forceinline__ v8f mma(v8f acc, v16bf a, v16bf b)
{
    return __builtin_amdgcn_wmma_f32_16x16x32_bf16(false, a, false, b,
                                                   (short)0, acc, false, false);
}
#define ZACC {0.f, 0.f, 0.f, 0.f, 0.f, 0.f, 0.f, 0.f}

// ---------- small utility kernels ----------
__global__ void zero_u16_kernel(u16* p, long n)
{
    long i = (long)blockIdx.x * blockDim.x + threadIdx.x;
    if (i < n) p[i] = 0;
}
__global__ void zero_f32_kernel(float* p, long n)
{
    long i = (long)blockIdx.x * blockDim.x + threadIdx.x;
    if (i < n) p[i] = 0.f;
}
__global__ void cvt_bf16_kernel(const float* __restrict__ s, u16* __restrict__ d,
                                long cols, long ldd, long coff, long n)
{
    long i = (long)blockIdx.x * blockDim.x + threadIdx.x;
    if (i >= n) return;
    long r = i / cols, c = i % cols;
    d[r * ldd + coff + c] = f2bf(s[i]);
}
// conv weight repack: src [Cout,Cin,KW] f32 -> dst [KW,Cout,CinP] bf16
__global__ void cvt_convw_kernel(const float* __restrict__ s, u16* __restrict__ d,
                                 int Cout, int Cin, int CinP, int KW)
{
    long n = (long)Cout * Cin * KW;
    long i = (long)blockIdx.x * blockDim.x + threadIdx.x;
    if (i >= n) return;
    int kw = (int)(i % KW);
    int ci = (int)((i / KW) % Cin);
    int co = (int)(i / ((long)KW * Cin));
    d[((long)kw * Cout + co) * CinP + ci] = f2bf(s[i]);
}
__global__ void vec_add_kernel(const float* a, const float* b, float* o, int n)
{
    int i = blockIdx.x * blockDim.x + threadIdx.x;
    if (i < n) o[i] = a[i] + b[i];
}
__global__ void copy_f32_kernel(const float* a, float* o, int n)
{
    int i = blockIdx.x * blockDim.x + threadIdx.x;
    if (i < n) o[i] = a[i];
}

// embedding gather -> [B, THE, 512] bf16 (time-major, halo offset HAL)
__global__ void embed_kernel(const int* __restrict__ text, const float* __restrict__ emb,
                             u16* __restrict__ X)
{
    long i = (long)blockIdx.x * blockDim.x + threadIdx.x;
    if (i >= (long)BB * TE * 512) return;
    int c = (int)(i % 512);
    int t = (int)((i / 512) % TE);
    int b = (int)(i / (512 * TE));
    int id = text[b * TE + t];
    X[((long)b * THE + HAL + t) * 512 + c] = f2bf(emb[(long)id * 512 + c]);
}

// [B, THE, 512] bf16 -> xs [TE*B, 512] bf16
__global__ void make_xs_kernel(const u16* __restrict__ cin, u16* __restrict__ xs)
{
    long i = (long)blockIdx.x * blockDim.x + threadIdx.x;
    if (i >= (long)TE * BB * 512) return;
    int c = (int)(i % 512);
    int b = (int)((i / 512) % BB);
    int t = (int)(i / (512 * BB));
    xs[i] = cin[((long)b * THE + HAL + t) * 512 + c];
}

// ---------- GEMM: C[M,N] = act(A[M,K] @ W[N,K]^T + bias) ----------
// requires M%16==0, N%64==0, K%32==0 (all launches padded to this)
// one wave per block; 16 x 64 per wave (A fragment reused 4x per k-step)
__global__ void gemm_kernel(const u16* __restrict__ A, int lda,
                            const u16* __restrict__ W, int ldw,
                            const float* __restrict__ bias,
                            float* __restrict__ C, int ldc,
                            u16* __restrict__ Cbf, int ldcb,
                            int K, int act)
{
    int m0 = blockIdx.x * 16, n0 = blockIdx.y * 64;
    const u16* Ap = A + (long)m0 * lda;
    v8f acc[4];
#pragma unroll
    for (int j = 0; j < 4; ++j) acc[j] = (v8f)ZACC;
    for (int k0 = 0; k0 < K; k0 += 32) {
        v16bf af = ldfa(Ap + k0, lda);
#pragma unroll
        for (int j = 0; j < 4; ++j)
            acc[j] = mma(acc[j], af, ldfb(W + (long)(n0 + 16 * j) * ldw + k0, ldw));
    }
    int lane = threadIdx.x & 31, grp = lane >> 4, nr = lane & 15;
#pragma unroll
    for (int j = 0; j < 4; ++j) {
#pragma unroll
        for (int r = 0; r < 8; ++r) {
            int m = m0 + grp * 8 + r, n = n0 + 16 * j + nr;
            float v = acc[j][r];
            if (bias) v += bias[n];
            v = actf(v, act);
            if (C)   C[(long)m * ldc + n] = v;
            if (Cbf) Cbf[(long)m * ldcb + n] = f2bf(v);
        }
    }
}

// ---------- conv1d (+bias +BN +act), activations [B, TH, C] with zero halo ----------
// weights pre-packed [KW, Cout, Cin]; one wave per block, 64(t) x 16(co)
__global__ void conv1d_kernel(const u16* __restrict__ X, const u16* __restrict__ Wp,
                              const float* __restrict__ bias, const float* __restrict__ g,
                              const float* __restrict__ be,
                              float* __restrict__ Y, u16* __restrict__ Ybf,
                              int Cin, int Cout, int T, int THx, int THy, int KW, int act)
{
    int t0 = blockIdx.x * 64, co0 = blockIdx.y * 16, b = blockIdx.z;
    int pad = (KW - 1) / 2;
    const u16* Xb = X + ((long)b * THx + HAL) * Cin;
    v8f acc[4];
#pragma unroll
    for (int j = 0; j < 4; ++j) acc[j] = (v8f)ZACC;
    for (int kw = 0; kw < KW; ++kw) {
        for (int c0 = 0; c0 < Cin; c0 += 32) {
            v16bf bf_ = ldfb(Wp + ((long)kw * Cout + co0) * Cin + c0, Cin);
#pragma unroll
            for (int j = 0; j < 4; ++j) {
                if (t0 + 16 * j >= T + pad) continue;   // tile fully past the end
                int tb = t0 + 16 * j + kw - pad;        // halo absorbs [-pad, T+pad)
                acc[j] = mma(acc[j], ldfa(Xb + (long)tb * Cin + c0, Cin), bf_);
            }
        }
    }
    float inv = rsqrtf(1.0f + 1e-5f);
    int lane = threadIdx.x & 31, grp = lane >> 4, nr = lane & 15;
#pragma unroll
    for (int j = 0; j < 4; ++j) {
#pragma unroll
        for (int r = 0; r < 8; ++r) {
            int t = t0 + 16 * j + grp * 8 + r, co = co0 + nr;
            if (t < T) {
                float v = acc[j][r] + bias[co];
                v = g[co] * v * inv + be[co];
                v = actf(v, act);
                long o = ((long)b * THy + HAL + t) * Cout + co;
                if (Y)   Y[o] = v;
                if (Ybf) Ybf[o] = f2bf(v);
            }
        }
    }
}

// ---------- persistent BiLSTM (grid = 2 blocks, one per direction) ----------
__global__ void lstm_kernel(const float* __restrict__ gF, const float* __restrict__ gB,
                            const u16* __restrict__ WhF, const u16* __restrict__ WhB,
                            float* __restrict__ enc, u16* __restrict__ encbf)
{
    __shared__ u16   hbf[BB * 256];     // 16 KB
    __shared__ float csh[BB * 256];     // 32 KB
    __shared__ float zsh[BB * 1024];    // 128 KB  (CDNA5 WGP: 320 KB LDS)
    int dir = blockIdx.x;
    const float* gates = dir ? gB : gF;
    const u16* Whh = dir ? WhB : WhF;
    for (int i = threadIdx.x; i < BB * 256; i += 1024) { hbf[i] = 0; csh[i] = 0.f; }
    __syncthreads();
    int wid = threadIdx.x >> 5;
    int lane = threadIdx.x & 31, grp = lane >> 4, nr = lane & 15;
    int tm = wid & 1, tnb = (wid >> 1) * 4;
    for (int s = 0; s < TE; ++s) {
        int t = dir ? (TE - 1 - s) : s;
        {
            v8f acc[4];
#pragma unroll
            for (int j = 0; j < 4; ++j) acc[j] = (v8f)ZACC;
            const u16* Ap = hbf + tm * 16 * 256;
            for (int k0 = 0; k0 < 256; k0 += 32) {
                v16bf af = ldfa(Ap + k0, 256);
#pragma unroll
                for (int j = 0; j < 4; ++j)
                    acc[j] = mma(acc[j], af, ldfb(Whh + (long)(tnb + j) * 16 * 256 + k0, 256));
            }
#pragma unroll
            for (int j = 0; j < 4; ++j) {
#pragma unroll
                for (int r = 0; r < 8; ++r) {
                    int m = tm * 16 + grp * 8 + r, n = (tnb + j) * 16 + nr;
                    zsh[m * 1024 + n] = acc[j][r] + gates[((long)t * BB + m) * 1024 + n];
                }
            }
        }
        __syncthreads();
        for (int i = threadIdx.x; i < BB * 256; i += 1024) {
            int b = i >> 8, j = i & 255;
            float zi = zsh[b * 1024 + j];
            float zf = zsh[b * 1024 + 256 + j];
            float zg = zsh[b * 1024 + 512 + j];
            float zo = zsh[b * 1024 + 768 + j];
            float c = sigf(zf) * csh[i] + sigf(zi) * tanhf(zg);
            float h = sigf(zo) * tanhf(c);
            csh[i] = c;
            hbf[i] = f2bf(h);
            long o = ((long)b * TE + t) * 512 + dir * 256 + j;
            enc[o] = h;
            encbf[o] = f2bf(h);
        }
        __syncthreads();
    }
}

// ---------- decoder helpers ----------
__global__ void make_decin_kernel(const float* __restrict__ mels, u16* __restrict__ decin)
{
    long i = (long)blockIdx.x * blockDim.x + threadIdx.x;
    if (i >= (long)NSTEP * BB * 96) return;
    int m = (int)(i % 96);
    int b = (int)((i / 96) % BB);
    int t = (int)(i / (96 * BB));
    float v = (m < NMEL && t > 0) ? mels[((long)b * TD + (2 * t - 1)) * NMEL + m] : 0.f;
    decin[i] = f2bf(v);
}

__global__ void dec_init_kernel(u16* xA, u16* xD, float* ac, float* dc,
                                float* aw, float* acum, unsigned int* bar)
{
    int i = blockIdx.x * blockDim.x + threadIdx.x;
    if (i < BB * 1792) xA[i] = 0;
    if (i < BB * 2560) xD[i] = 0;
    if (i < BB * 1024) { ac[i] = 0.f; dc[i] = 0.f; }
    if (i < BB * TE) { aw[i] = 0.f; acum[i] = 0.f; }
    if (i < 2) bar[i] = 0u;
}

__device__ __forceinline__ void gbar(unsigned int* bar)
{
    __syncthreads();
    if (threadIdx.x == 0) {
        __threadfence();
        unsigned int gen = __hip_atomic_load(bar + 1, __ATOMIC_ACQUIRE, __HIP_MEMORY_SCOPE_AGENT);
        unsigned int old = __hip_atomic_fetch_add(bar, 1u, __ATOMIC_ACQ_REL, __HIP_MEMORY_SCOPE_AGENT);
        if (old == (unsigned)NB - 1u) {
            __hip_atomic_store(bar, 0u, __ATOMIC_RELAXED, __HIP_MEMORY_SCOPE_AGENT);
            __hip_atomic_fetch_add(bar + 1, 1u, __ATOMIC_RELEASE, __HIP_MEMORY_SCOPE_AGENT);
        } else {
            while (__hip_atomic_load(bar + 1, __ATOMIC_ACQUIRE, __HIP_MEMORY_SCOPE_AGENT) == gen)
                __builtin_amdgcn_s_sleep(8);
        }
    }
    __syncthreads();
}

struct DecPtrs {
    const u16* pre_all;   // [NSTEP*32, 256]
    const u16* WcatA;     // [4096, 1792] = [Wih(768) | Whh(1024)]
    const float* bA;      // [4096]
    const u16* WcatD;     // [4096, 2560]
    const float* bD;
    const u16* Wq;        // [128, 1024]
    const float* pmem;    // [32*TE, 128]
    const float* enc;     // [32*TE, 512]
    const float* locW;    // [128, 32]
    const float* locC;    // [32, 2, 31]
    const float* attv;    // [128]
    const int* lens;
    float* xq;            // [32,128]
    u16* xcatA;           // [32,1792] = [pre | ctx | ah]
    u16* xcatD;           // [32,2560] = [ah | ctx | dh]
    float* zA;            // [32,4096]
    float* zD;
    float* ac;            // [32,1024]
    float* dc;
    float* aw;            // [32,TE]
    float* acum;
    float* evec;
    float* locf;          // [32*TE, 32]
    u16* proj_in;         // [NSTEP*32, 1536] = [dh | ctx]
    float* align_out;     // [32, NSTEP, TE]
    unsigned int* bar;
};

__global__ void __launch_bounds__(TPB) decoder_kernel(DecPtrs P)
{
    const int NT = NB * TPB;
    int gid = blockIdx.x * TPB + threadIdx.x;
    int gw = gid >> 5;                          // global wave 0..127
    int lane = threadIdx.x & 31, grp = lane >> 4, nr = lane & 15;
    int tm = gw & 1, tnb = (gw >> 1) * 4;       // wave -> 16(m) x 64(n) GEMM job
    for (int t = 0; t < NSTEP; ++t) {
        // A: prenet output of this step -> xcatA[:, 0:256]
        for (int i = gid; i < BB * 256; i += NT) {
            int b = i >> 8, j = i & 255;
            P.xcatA[b * 1792 + j] = P.pre_all[((long)t * BB + b) * 256 + j];
        }
        gbar(P.bar);
        // B: zA = xcatA @ WcatA^T + bA   (K=1792; 128 wave-jobs of 16x64)
        {
            v8f acc[4];
#pragma unroll
            for (int j = 0; j < 4; ++j) acc[j] = (v8f)ZACC;
            const u16* Ap = P.xcatA + (long)tm * 16 * 1792;
            for (int k0 = 0; k0 < 1792; k0 += 32) {
                v16bf af = ldfa(Ap + k0, 1792);
#pragma unroll
                for (int j = 0; j < 4; ++j)
                    acc[j] = mma(acc[j], af,
                                 ldfb(P.WcatA + (long)(tnb + j) * 16 * 1792 + k0, 1792));
            }
#pragma unroll
            for (int j = 0; j < 4; ++j)
#pragma unroll
                for (int r = 0; r < 8; ++r) {
                    int m = tm * 16 + grp * 8 + r, n = (tnb + j) * 16 + nr;
                    P.zA[m * 4096 + n] = acc[j][r] + P.bA[n];
                }
        }
        gbar(P.bar);
        // C: attention-LSTM pointwise
        for (int i = gid; i < BB * 1024; i += NT) {
            int b = i >> 10, j = i & 1023;
            float zi = P.zA[b * 4096 + j], zf = P.zA[b * 4096 + 1024 + j];
            float zg = P.zA[b * 4096 + 2048 + j], zo = P.zA[b * 4096 + 3072 + j];
            float c = sigf(zf) * P.ac[i] + sigf(zi) * tanhf(zg);
            float h = sigf(zo) * tanhf(c);
            P.ac[i] = c;
            u16 hb = f2bf(h);
            P.xcatA[b * 1792 + 768 + j] = hb;
            P.xcatD[b * 2560 + j] = hb;
        }
        gbar(P.bar);
        // D: q = ah @ Wq^T (waves 0..3, 16x64 each) + location conv (all threads)
        if (gw < 4) {
            v8f acc[4];
#pragma unroll
            for (int j = 0; j < 4; ++j) acc[j] = (v8f)ZACC;
            const u16* Ap = P.xcatA + (long)tm * 16 * 1792 + 768;
            for (int k0 = 0; k0 < 1024; k0 += 32) {
                v16bf af = ldfa(Ap + k0, 1792);
#pragma unroll
                for (int j = 0; j < 4; ++j)
                    acc[j] = mma(acc[j], af,
                                 ldfb(P.Wq + (long)(tnb + j) * 16 * 1024 + k0, 1024));
            }
#pragma unroll
            for (int j = 0; j < 4; ++j)
#pragma unroll
                for (int r = 0; r < 8; ++r)
                    P.xq[(tm * 16 + grp * 8 + r) * 128 + (tnb + j) * 16 + nr] = acc[j][r];
        }
        for (int i = gid; i < BB * TE * 32; i += NT) {
            int c = i & 31, tt = (i >> 5) % TE, b = i / (TE * 32);
            float s = 0.f;
            for (int k = 0; k < 31; ++k) {
                int tp = tt + k - 15;
                if (tp >= 0 && tp < TE) {
                    s += P.locC[(c * 2 + 0) * 31 + k] * P.aw[b * TE + tp];
                    s += P.locC[(c * 2 + 1) * 31 + k] * P.acum[b * TE + tp];
                }
            }
            P.locf[((long)b * TE + tt) * 32 + c] = s;
        }
        gbar(P.bar);
        // E: energies  e[b,tt] = v . tanh(q + pmem + locW @ locf)
        for (int row = gw; row < BB * TE; row += NT / 32) {
            int b = row / TE, tt = row % TE;
            const float* lf = P.locf + (long)row * 32;
            float part = 0.f;
            for (int d = lane; d < 128; d += 32) {
                float lp = 0.f;
                for (int c = 0; c < 32; ++c) lp += P.locW[d * 32 + c] * lf[c];
                part += P.attv[d] * tanhf(P.xq[b * 128 + d] + P.pmem[(long)row * 128 + d] + lp);
            }
            for (int o = 16; o > 0; o >>= 1) part += __shfl_down(part, o, 32);
            if (lane == 0)
                P.evec[row] = (tt < P.lens[b]) ? part : -1e9f;
        }
        gbar(P.bar);
        // F: softmax + acum + alignments (one wave per batch row)
        if (gw < BB) {
            int b = gw;
            float mx = -1e30f;
            for (int tt = lane; tt < TE; tt += 32) mx = fmaxf(mx, P.evec[b * TE + tt]);
            for (int o = 16; o > 0; o >>= 1) mx = fmaxf(mx, __shfl_xor(mx, o, 32));
            float sm = 0.f;
            for (int tt = lane; tt < TE; tt += 32) sm += __expf(P.evec[b * TE + tt] - mx);
            for (int o = 16; o > 0; o >>= 1) sm += __shfl_xor(sm, o, 32);
            float inv = 1.f / sm;
            for (int tt = lane; tt < TE; tt += 32) {
                float a = __expf(P.evec[b * TE + tt] - mx) * inv;
                P.aw[b * TE + tt] = a;
                P.acum[b * TE + tt] += a;
                P.align_out[((long)b * NSTEP + t) * TE + tt] = a;
            }
        }
        gbar(P.bar);
        // G: context = aw @ enc
        for (int i = gid; i < BB * 512; i += NT) {
            int b = i >> 9, d = i & 511;
            float s = 0.f;
            for (int tt = 0; tt < TE; ++tt)
                s += P.aw[b * TE + tt] * P.enc[((long)b * TE + tt) * 512 + d];
            u16 cb = f2bf(s);
            P.xcatA[b * 1792 + 256 + d] = cb;
            P.xcatD[b * 2560 + 1024 + d] = cb;
            P.proj_in[((long)t * BB + b) * 1536 + 1024 + d] = cb;
        }
        gbar(P.bar);
        // H: zD = xcatD @ WcatD^T + bD   (K=2560; 128 wave-jobs of 16x64)
        {
            v8f acc[4];
#pragma unroll
            for (int j = 0; j < 4; ++j) acc[j] = (v8f)ZACC;
            const u16* Ap = P.xcatD + (long)tm * 16 * 2560;
            for (int k0 = 0; k0 < 2560; k0 += 32) {
                v16bf af = ldfa(Ap + k0, 2560);
#pragma unroll
                for (int j = 0; j < 4; ++j)
                    acc[j] = mma(acc[j], af,
                                 ldfb(P.WcatD + (long)(tnb + j) * 16 * 2560 + k0, 2560));
            }
#pragma unroll
            for (int j = 0; j < 4; ++j)
#pragma unroll
                for (int r = 0; r < 8; ++r) {
                    int m = tm * 16 + grp * 8 + r, n = (tnb + j) * 16 + nr;
                    P.zD[m * 4096 + n] = acc[j][r] + P.bD[n];
                }
        }
        gbar(P.bar);
        // I: decoder-LSTM pointwise
        for (int i = gid; i < BB * 1024; i += NT) {
            int b = i >> 10, j = i & 1023;
            float zi = P.zD[b * 4096 + j], zf = P.zD[b * 4096 + 1024 + j];
            float zg = P.zD[b * 4096 + 2048 + j], zo = P.zD[b * 4096 + 3072 + j];
            float c = sigf(zf) * P.dc[i] + sigf(zi) * tanhf(zg);
            float h = sigf(zo) * tanhf(c);
            P.dc[i] = c;
            u16 hb = f2bf(h);
            P.xcatD[b * 2560 + 1536 + j] = hb;
            P.proj_in[((long)t * BB + b) * 1536 + j] = hb;
        }
        gbar(P.bar);
    }
}

// ---------- mel/stop scatter + postnet finish ----------
__global__ void scatter_kernel(const float* __restrict__ ms, float* __restrict__ mel_out,
                               float* __restrict__ stop_out, u16* __restrict__ melbf)
{
    long n1 = (long)NSTEP * BB * 160;
    long i = (long)blockIdx.x * blockDim.x + threadIdx.x;
    if (i < n1) {
        int j = (int)(i % 160);
        int b = (int)((i / 160) % BB);
        int s = (int)(i / (160 * BB));
        int rr = j / NMEL, m = j % NMEL;
        int tt = s * 2 + rr;
        float v = ms[((long)s * BB + b) * 192 + j];
        mel_out[((long)b * TD + tt) * NMEL + m] = v;
        melbf[((long)b * THP + HAL + tt) * 96 + m] = f2bf(v);
    } else if (i < n1 + (long)NSTEP * BB) {
        long k = i - n1;
        int b = (int)(k % BB);
        int s = (int)(k / BB);
        float v = sigf(ms[((long)s * BB + b) * 192 + 160]);
        stop_out[b * TD + 2 * s] = v;
        stop_out[b * TD + 2 * s + 1] = v;
    }
}

__global__ void post_add_kernel(const float* __restrict__ mel_out,
                                const float* __restrict__ pY, float* __restrict__ post_out)
{
    long i = (long)blockIdx.x * blockDim.x + threadIdx.x;
    if (i >= (long)BB * TD * NMEL) return;
    int m = (int)(i % NMEL);
    int tt = (int)((i / NMEL) % TD);
    int b = (int)(i / (NMEL * TD));
    post_out[i] = mel_out[i] + pY[((long)b * THP + HAL + tt) * NMEL + m];
}

// =====================================================================
extern "C" void kernel_launch(void* const* d_in, const int* in_sizes, int n_in,
                              void* d_out, int out_size, void* d_ws, size_t ws_size,
                              hipStream_t stream)
{
    (void)in_sizes; (void)n_in; (void)out_size; (void)ws_size;
    const int* text = (const int*)d_in[0];
    const int* lens = (const int*)d_in[1];
    const float* mels = (const float*)d_in[2];
    const float* embW = (const float*)d_in[3];
    const float *ecW[3], *ecB[3], *ecG[3], *ecBe[3];
    for (int i = 0; i < 3; ++i) {
        ecW[i] = (const float*)d_in[4 + 4 * i]; ecB[i] = (const float*)d_in[5 + 4 * i];
        ecG[i] = (const float*)d_in[6 + 4 * i]; ecBe[i] = (const float*)d_in[7 + 4 * i];
    }
    const float* lfWih = (const float*)d_in[16]; const float* lfWhh = (const float*)d_in[17];
    const float* lfbih = (const float*)d_in[18]; const float* lfbhh = (const float*)d_in[19];
    const float* lbWih = (const float*)d_in[20]; const float* lbWhh = (const float*)d_in[21];
    const float* lbbih = (const float*)d_in[22]; const float* lbbhh = (const float*)d_in[23];
    const float* memW = (const float*)d_in[24];
    const float* p1W = (const float*)d_in[25]; const float* p1b = (const float*)d_in[26];
    const float* p2W = (const float*)d_in[27]; const float* p2b = (const float*)d_in[28];
    const float* aWih = (const float*)d_in[29]; const float* aWhh = (const float*)d_in[30];
    const float* abih = (const float*)d_in[31]; const float* abhh = (const float*)d_in[32];
    const float* qW = (const float*)d_in[33];
    const float* locC = (const float*)d_in[34];
    const float* locW = (const float*)d_in[35];
    const float* attv = (const float*)d_in[36];
    const float* dWih = (const float*)d_in[37]; const float* dWhh = (const float*)d_in[38];
    const float* dbih = (const float*)d_in[39]; const float* dbhh = (const float*)d_in[40];
    const float* melW = (const float*)d_in[41]; const float* melb = (const float*)d_in[42];
    const float* stopW = (const float*)d_in[43]; const float* stopb = (const float*)d_in[44];
    const float *pWc[5], *pBc[5], *pGc[5], *pBec[5];
    for (int i = 0; i < 5; ++i) {
        pWc[i] = (const float*)d_in[45 + 4 * i]; pBc[i] = (const float*)d_in[46 + 4 * i];
        pGc[i] = (const float*)d_in[47 + 4 * i]; pBec[i] = (const float*)d_in[48 + 4 * i];
    }
    float* out_mel = (float*)d_out;             // [32,800,80]
    float* out_post = out_mel + 2048000;        // [32,800,80]
    float* out_stop = out_post + 2048000;       // [32,800]
    float* out_align = out_stop + 25600;        // [32,400,200]

    char* wsp = (char*)d_ws; size_t off = 0;
    auto alloc = [&](size_t bytes) -> void* {
        void* p = wsp + off; off += (bytes + 255) & ~(size_t)255; return p;
    };
    // conv activations: [B, TH, C] bf16 with zero halo
    u16* Xbf    = (u16*)alloc((size_t)BB * THE * 512 * 2);
    u16* cbuf0  = (u16*)alloc((size_t)BB * THE * 512 * 2);
    u16* cbuf1  = (u16*)alloc((size_t)BB * THE * 512 * 2);
    u16* ecWp[3]; for (int i = 0; i < 3; ++i) ecWp[i] = (u16*)alloc((size_t)5 * 512 * 512 * 2);
    u16* xsbf   = (u16*)alloc((size_t)TE * BB * 512 * 2);
    u16* WihFbf = (u16*)alloc((size_t)1024 * 512 * 2);
    u16* WihBbf = (u16*)alloc((size_t)1024 * 512 * 2);
    u16* WhhFbf = (u16*)alloc((size_t)1024 * 256 * 2);
    u16* WhhBbf = (u16*)alloc((size_t)1024 * 256 * 2);
    float* bsumF = (float*)alloc(1024 * 4);
    float* bsumB = (float*)alloc(1024 * 4);
    float* gatesF = (float*)alloc((size_t)TE * BB * 1024 * 4);
    float* gatesB = (float*)alloc((size_t)TE * BB * 1024 * 4);
    float* enc   = (float*)alloc((size_t)BB * TE * 512 * 4);
    u16* encbf   = (u16*)alloc((size_t)BB * TE * 512 * 2);
    u16* memWbf  = (u16*)alloc((size_t)128 * 512 * 2);
    float* pmem  = (float*)alloc((size_t)BB * TE * 128 * 4);
    u16* decin   = (u16*)alloc((size_t)NSTEP * BB * 96 * 2);      // K padded 80->96
    u16* p1Wbf   = (u16*)alloc((size_t)256 * 96 * 2);
    u16* p2Wbf   = (u16*)alloc((size_t)256 * 256 * 2);
    u16* pretmp  = (u16*)alloc((size_t)NSTEP * BB * 256 * 2);
    u16* pre_all = (u16*)alloc((size_t)NSTEP * BB * 256 * 2);
    u16* WcatA   = (u16*)alloc((size_t)4096 * 1792 * 2);
    float* bA    = (float*)alloc(4096 * 4);
    u16* WcatD   = (u16*)alloc((size_t)4096 * 2560 * 2);
    float* bD    = (float*)alloc(4096 * 4);
    u16* Wqbf    = (u16*)alloc((size_t)128 * 1024 * 2);
    u16* Wms     = (u16*)alloc((size_t)192 * 1536 * 2);           // N padded 161->192
    float* bms   = (float*)alloc(192 * 4);
    float* xq    = (float*)alloc((size_t)BB * 128 * 4);
    u16* xcatA   = (u16*)alloc((size_t)BB * 1792 * 2);
    u16* xcatD   = (u16*)alloc((size_t)BB * 2560 * 2);
    float* zA    = (float*)alloc((size_t)BB * 4096 * 4);
    float* zD    = (float*)alloc((size_t)BB * 4096 * 4);
    float* acb   = (float*)alloc((size_t)BB * 1024 * 4);
    float* dcb   = (float*)alloc((size_t)BB * 1024 * 4);
    float* aw    = (float*)alloc((size_t)BB * TE * 4);
    float* acum  = (float*)alloc((size_t)BB * TE * 4);
    float* evec  = (float*)alloc((size_t)BB * TE * 4);
    float* locf  = (float*)alloc((size_t)BB * TE * 32 * 4);
    u16* proj_in = (u16*)alloc((size_t)NSTEP * BB * 1536 * 2);
    float* melstop = (float*)alloc((size_t)NSTEP * BB * 192 * 4);
    u16* melbf   = (u16*)alloc((size_t)BB * THP * 96 * 2);        // Cin padded 80->96
    u16* postWp[5];
    int pcin[5]  = {80, 512, 512, 512, 512};
    int pcinp[5] = {96, 512, 512, 512, 512};
    int pcout[5] = {512, 512, 512, 512, 80};
    for (int i = 0; i < 5; ++i)
        postWp[i] = (u16*)alloc((size_t)5 * pcout[i] * pcinp[i] * 2);
    u16* pbuf0 = (u16*)alloc((size_t)BB * THP * 512 * 2);
    u16* pbuf1 = (u16*)alloc((size_t)BB * THP * 512 * 2);
    float* pY  = (float*)alloc((size_t)BB * THP * 80 * 4);
    unsigned int* bar = (unsigned int*)alloc(256);

    auto zu16 = [&](u16* p, long n) {
        zero_u16_kernel<<<dim3((unsigned)((n + 255) / 256)), dim3(256), 0, stream>>>(p, n);
    };
    auto cvt = [&](const float* s, u16* dst, long rows, long cols, long ldd, long coff) {
        long n = rows * cols;
        cvt_bf16_kernel<<<dim3((unsigned)((n + 255) / 256)), dim3(256), 0, stream>>>(
            s, dst, cols, ldd, coff, n);
    };
    auto cvtw = [&](const float* s, u16* dst, int Cout, int Cin, int CinP) {
        long n = (long)Cout * Cin * 5;
        cvt_convw_kernel<<<dim3((unsigned)((n + 255) / 256)), dim3(256), 0, stream>>>(
            s, dst, Cout, Cin, CinP, 5);
    };
    auto gemm = [&](const u16* A, int lda, const u16* W, int ldw, const float* bias,
                    float* C, int ldc, u16* Cbf, int ldcb, int M, int N, int K, int act) {
        gemm_kernel<<<dim3(M / 16, N / 64), dim3(32), 0, stream>>>(
            A, lda, W, ldw, bias, C, ldc, Cbf, ldcb, K, act);
    };
    auto conv = [&](const u16* X, const u16* W, const float* b, const float* g,
                    const float* be, float* Y, u16* Ybf, int Cin, int Cout, int T,
                    int THx, int THy, int act) {
        conv1d_kernel<<<dim3((T + 63) / 64, Cout / 16, BB), dim3(32), 0, stream>>>(
            X, W, b, g, be, Y, Ybf, Cin, Cout, T, THx, THy, 5, act);
    };

    // ---- zero-fill halos / padded regions (every call; buffers are poisoned once) ----
    zu16(Xbf,   (long)BB * THE * 512);
    zu16(cbuf0, (long)BB * THE * 512);
    zu16(cbuf1, (long)BB * THE * 512);
    zu16(melbf, (long)BB * THP * 96);
    zu16(pbuf0, (long)BB * THP * 512);
    zu16(pbuf1, (long)BB * THP * 512);
    zu16(p1Wbf, 256 * 96);
    zu16(Wms,   (long)192 * 1536);
    zu16(postWp[0], (long)5 * 512 * 96);
    zero_f32_kernel<<<dim3(1), dim3(256), 0, stream>>>(bms, 192);

    // ---- weight packing (f32 -> bf16, concatenated / per-tap layouts) ----
    for (int i = 0; i < 3; ++i) cvtw(ecW[i], ecWp[i], 512, 512, 512);
    for (int i = 0; i < 5; ++i) cvtw(pWc[i], postWp[i], pcout[i], pcin[i], pcinp[i]);
    cvt(lfWih, WihFbf, 1024, 512, 512, 0);  cvt(lbWih, WihBbf, 1024, 512, 512, 0);
    cvt(lfWhh, WhhFbf, 1024, 256, 256, 0);  cvt(lbWhh, WhhBbf, 1024, 256, 256, 0);
    cvt(memW, memWbf, 128, 512, 512, 0);
    cvt(p1W, p1Wbf, 256, 80, 96, 0);        cvt(p2W, p2Wbf, 256, 256, 256, 0);
    cvt(aWih, WcatA, 4096, 768, 1792, 0);   cvt(aWhh, WcatA, 4096, 1024, 1792, 768);
    cvt(dWih, WcatD, 4096, 1536, 2560, 0);  cvt(dWhh, WcatD, 4096, 1024, 2560, 1536);
    cvt(qW, Wqbf, 128, 1024, 1024, 0);
    cvt(melW, Wms, 160, 1536, 1536, 0);     cvt(stopW, Wms + (size_t)160 * 1536, 1, 1536, 1536, 0);
    vec_add_kernel<<<dim3(4), dim3(256), 0, stream>>>(lfbih, lfbhh, bsumF, 1024);
    vec_add_kernel<<<dim3(4), dim3(256), 0, stream>>>(lbbih, lbbhh, bsumB, 1024);
    vec_add_kernel<<<dim3(16), dim3(256), 0, stream>>>(abih, abhh, bA, 4096);
    vec_add_kernel<<<dim3(16), dim3(256), 0, stream>>>(dbih, dbhh, bD, 4096);
    copy_f32_kernel<<<dim3(1), dim3(256), 0, stream>>>(melb, bms, 160);
    copy_f32_kernel<<<dim3(1), dim3(32), 0, stream>>>(stopb, bms + 160, 1);

    // ---- encoder ----
    embed_kernel<<<dim3((BB * TE * 512 + 255) / 256), dim3(256), 0, stream>>>(text, embW, Xbf);
    conv(Xbf,   ecWp[0], ecB[0], ecG[0], ecBe[0], nullptr, cbuf0, 512, 512, TE, THE, THE, 1);
    conv(cbuf0, ecWp[1], ecB[1], ecG[1], ecBe[1], nullptr, cbuf1, 512, 512, TE, THE, THE, 1);
    conv(cbuf1, ecWp[2], ecB[2], ecG[2], ecBe[2], nullptr, cbuf0, 512, 512, TE, THE, THE, 1);
    make_xs_kernel<<<dim3((TE * BB * 512 + 255) / 256), dim3(256), 0, stream>>>(cbuf0, xsbf);
    gemm(xsbf, 512, WihFbf, 512, bsumF, gatesF, 1024, nullptr, 0, TE * BB, 1024, 512, 0);
    gemm(xsbf, 512, WihBbf, 512, bsumB, gatesB, 1024, nullptr, 0, TE * BB, 1024, 512, 0);
    lstm_kernel<<<dim3(2), dim3(1024), 0, stream>>>(gatesF, gatesB, WhhFbf, WhhBbf, enc, encbf);
    gemm(encbf, 512, memWbf, 512, nullptr, pmem, 128, nullptr, 0, BB * TE, 128, 512, 0);

    // ---- decoder (prenet hoisted; persistent attention loop) ----
    make_decin_kernel<<<dim3((NSTEP * BB * 96 + 255) / 256), dim3(256), 0, stream>>>(mels, decin);
    gemm(decin, 96, p1Wbf, 96, p1b, nullptr, 0, pretmp, 256, NSTEP * BB, 256, 96, 1);
    gemm(pretmp, 256, p2Wbf, 256, p2b, nullptr, 0, pre_all, 256, NSTEP * BB, 256, 256, 1);
    dec_init_kernel<<<dim3((BB * 2560 + 255) / 256), dim3(256), 0, stream>>>(
        xcatA, xcatD, acb, dcb, aw, acum, bar);
    DecPtrs P;
    P.pre_all = pre_all; P.WcatA = WcatA; P.bA = bA; P.WcatD = WcatD; P.bD = bD;
    P.Wq = Wqbf; P.pmem = pmem; P.enc = enc; P.locW = locW; P.locC = locC;
    P.attv = attv; P.lens = lens; P.xq = xq; P.xcatA = xcatA; P.xcatD = xcatD;
    P.zA = zA; P.zD = zD; P.ac = acb; P.dc = dcb; P.aw = aw; P.acum = acum;
    P.evec = evec; P.locf = locf; P.proj_in = proj_in; P.align_out = out_align; P.bar = bar;
    decoder_kernel<<<dim3(NB), dim3(TPB), 0, stream>>>(P);

    // ---- deferred projection + postnet ----
    gemm(proj_in, 1536, Wms, 1536, bms, melstop, 192, nullptr, 0, NSTEP * BB, 192, 1536, 0);
    scatter_kernel<<<dim3((unsigned)(((long)NSTEP * BB * 160 + NSTEP * BB + 255) / 256)),
                     dim3(256), 0, stream>>>(melstop, out_mel, out_stop, melbf);
    conv(melbf, postWp[0], pBc[0], pGc[0], pBec[0], nullptr, pbuf0, 96, 512, TD, THP, THP, 2);
    conv(pbuf0, postWp[1], pBc[1], pGc[1], pBec[1], nullptr, pbuf1, 512, 512, TD, THP, THP, 2);
    conv(pbuf1, postWp[2], pBc[2], pGc[2], pBec[2], nullptr, pbuf0, 512, 512, TD, THP, THP, 2);
    conv(pbuf0, postWp[3], pBc[3], pGc[3], pBec[3], nullptr, pbuf1, 512, 512, TD, THP, THP, 2);
    conv(pbuf1, postWp[4], pBc[4], pGc[4], pBec[4], pY, nullptr, 512, 80, TD, THP, THP, 0);
    post_add_kernel<<<dim3((BB * TD * 80 + 255) / 256), dim3(256), 0, stream>>>(
        out_mel, pY, out_post);
}
